// AverageNode2Vec_41566693490863
// MI455X (gfx1250) — matrix operand order repaired and checked by reading, not compiled, not run on hardware
//
#include <hip/hip_runtime.h>
#include <math.h>

// ---------------- problem constants (match reference) ----------------
#define VOCAB 1000000
#define D     128
#define BATCH 8192
#define NEG   5
#define L     10

#define TILE  16            // examples per block (one WMMA tile edge)
#define ROWS  (TILE + TILE + TILE*NEG)   // 16 u + 16 v + 80 neg = 112
#define STR   132           // padded LDS row stride in floats (132*4B = 528B, 16B aligned, kills bank conflicts)
#define NWAVE 8             // 256 threads / wave32

typedef __attribute__((ext_vector_type(16))) _Float16 v16h;
typedef __attribute__((ext_vector_type(8)))  float    v8f;

__device__ __forceinline__ float log_sigmoid(float x) {
    // stable: min(x,0) - log1p(exp(-|x|))
    return fminf(x, 0.0f) - log1pf(expf(-fabsf(x)));
}

__global__ __launch_bounds__(256)
void n2v_fused(const float* __restrict__ u_table, const float* __restrict__ v_table,
               const int* __restrict__ pos_u, const int* __restrict__ pos_u_lens,
               const int* __restrict__ pos_v, const int* __restrict__ pos_v_lens,
               const int* __restrict__ neg_v, const int* __restrict__ neg_v_lens,
               float* __restrict__ block_partials) {
    __shared__ float smem[ROWS * STR];   // ~57.8 KB: averaged embedding rows
    __shared__ float loss_part[TILE];

    const int wave = threadIdx.x >> 5;
    const int lane = threadIdx.x & 31;
    const int b0   = blockIdx.x * TILE;

    // -------- Phase 1: gather + length-average 112 rows ------------------
    // One wave per row: lane l owns floats [4l, 4l+4) -> one global_load_b128
    // per (row, index); 512B per row, fully coalesced.
    for (int r = wave; r < ROWS; r += NWAVE) {
        const float* tab;
        const int*   idxp;
        int          len;
        if (r < TILE) {                       // emb_u rows
            const int gb = b0 + r;
            tab = u_table; idxp = pos_u + gb * L; len = pos_u_lens[gb];
        } else if (r < 2 * TILE) {            // emb_v rows
            const int gb = b0 + (r - TILE);
            tab = v_table; idxp = pos_v + gb * L; len = pos_v_lens[gb];
        } else {                              // neg rows: flat index b0*NEG + k
            const int gr = b0 * NEG + (r - 2 * TILE);
            tab = v_table; idxp = neg_v + gr * L; len = neg_v_lens[gr];
        }
        float ax = 0.f, ay = 0.f, az = 0.f, aw = 0.f;
        #pragma unroll
        for (int j = 0; j < L; ++j) {
            const int idx = idxp[j];          // uniform across wave -> scalar branch
            if (idx != 0) {                   // padding idx 0 contributes zero
                const float4 e = *(const float4*)(tab + (size_t)idx * D + lane * 4);
                ax += e.x; ay += e.y; az += e.z; aw += e.w;
            }
        }
        const float inv = 1.0f / (float)len;
        float* dst = &smem[r * STR + lane * 4];
        dst[0] = ax * inv; dst[1] = ay * inv; dst[2] = az * inv; dst[3] = aw * inv;
    }
    __syncthreads();

    // -------- Phase 2: 6 batched dot products via WMMA diagonals ---------
    // A = emb_u tile (16x128). Per target t, B column n = that example's own
    // target vector; only the 16x16 accumulator diagonal is the wanted score.
    if (wave == 0) {                          // full wave -> EXEC all ones for WMMA
        const int half = lane >> 4;
        const int m    = lane & 15;           // A row index == B column index
        // which lanes hold diagonal elements of the f32 C/D layout:
        //   lanes 0-7  -> M=N=lane   (vgpr lane)
        //   lanes 24-31-> M=N=lane-16 (vgpr lane-24)
        const bool has_diag = (lane < 8) || (lane >= 24);
        const int  dcol = (lane < 8) ? lane : (lane - 16);
        const int  dreg = (lane < 8) ? lane : (lane - 24);

        float lsum = 0.0f;
        #pragma unroll 1
        for (int t = 0; t < 6; ++t) {
            const float* brow = (t == 0)
                ? &smem[(TILE + m) * STR]                         // emb_v[b0+m]
                : &smem[(2 * TILE + m * NEG + (t - 1)) * STR];    // neg_emb[b0+m][t-1]
            v8f c = {};
            #pragma unroll
            for (int kc = 0; kc < D; kc += 32) {
                v16h a, b;
                #pragma unroll
                for (int v = 0; v < 8; ++v) {
                    // A 16x32 f16 layout (ISA 7.12.2): VGPR v, lane-half h:
                    //   K = (v>=4?16:0) + h*8 + (v&3)*2 , +1
                    const int ka = kc + ((v >= 4) ? 16 : 0) + half * 8 + (v & 3) * 2;
                    a[2 * v]     = (_Float16)smem[m * STR + ka];
                    a[2 * v + 1] = (_Float16)smem[m * STR + ka + 1];
                    // B 32x16 f16: same K order as A: VGPR v, lane-half h:
                    //   K = (v>=4?16:0) + h*8 + (v&3)*2 , +1
                    const int kb = kc + ((v >= 4) ? 16 : 0) + half * 8 + (v & 3) * 2;
                    b[2 * v]     = (_Float16)brow[kb];
                    b[2 * v + 1] = (_Float16)brow[kb + 1];
                }
                c = __builtin_amdgcn_wmma_f32_16x16x32_f16(false, a, false, b,
                                                           (short)0, c, false, false);
            }
            if (has_diag) {
                const float s = c[dreg];
                lsum += (t == 0) ? log_sigmoid(s) : log_sigmoid(-s);
            }
        }
        if (has_diag) loss_part[dcol] = lsum;
    }
    __syncthreads();

    if (threadIdx.x == 0) {
        float tot = 0.0f;
        #pragma unroll
        for (int i = 0; i < TILE; ++i) tot += loss_part[i];
        block_partials[blockIdx.x] = tot;
    }
}

// -------- final reduction over the 512 block partials -> scalar loss -----
__global__ __launch_bounds__(512)
void n2v_reduce(const float* __restrict__ partials, float* __restrict__ out) {
    __shared__ float red[512];
    const int tid = threadIdx.x;
    red[tid] = partials[tid];
    __syncthreads();
    for (int s = 256; s > 0; s >>= 1) {
        if (tid < s) red[tid] += red[tid + s];
        __syncthreads();
    }
    if (tid == 0) out[0] = -red[0] / (float)BATCH;
}

extern "C" void kernel_launch(void* const* d_in, const int* in_sizes, int n_in,
                              void* d_out, int out_size, void* d_ws, size_t ws_size,
                              hipStream_t stream) {
    (void)in_sizes; (void)n_in; (void)out_size; (void)ws_size;
    const float* u_table    = (const float*)d_in[0];
    const float* v_table    = (const float*)d_in[1];
    const int*   pos_u      = (const int*)d_in[2];
    const int*   pos_u_lens = (const int*)d_in[3];
    const int*   pos_v      = (const int*)d_in[4];
    const int*   pos_v_lens = (const int*)d_in[5];
    const int*   neg_v      = (const int*)d_in[6];
    const int*   neg_v_lens = (const int*)d_in[7];

    float* partials = (float*)d_ws;        // 512 floats of scratch
    float* out      = (float*)d_out;       // single f32 scalar

    const int nblocks = BATCH / TILE;      // 512
    hipLaunchKernelGGL(n2v_fused, dim3(nblocks), dim3(256), 0, stream,
                       u_table, v_table, pos_u, pos_u_lens,
                       pos_v, pos_v_lens, neg_v, neg_v_lens, partials);
    hipLaunchKernelGGL(n2v_reduce, dim3(1), dim3(512), 0, stream, partials, out);
}
